// SparseResidual3d_17136919511608
// MI455X (gfx1250) — matrix operand order repaired and checked by reading, not compiled, not run on hardware
//
#include <hip/hip_runtime.h>
#include <stdint.h>

// ---------------- problem constants ----------------
#define N_PTS   400000
#define CCH     64
#define K27     27
#define EPSV    1e-5f
#define LDA     72      // padded halves per LDS A row (144 B: 16B-aligned, bank-shifted)
#define MTILE   32      // points per block

typedef _Float16 h8v  __attribute__((ext_vector_type(8)));
typedef _Float16 v16h __attribute__((ext_vector_type(16)));
typedef float    v8f  __attribute__((ext_vector_type(8)));
typedef float    f4   __attribute__((ext_vector_type(4)));

// gcc-style int4 to match the async-to-LDS builtin's parameter type
typedef int i4g __attribute__((vector_size(16)));
typedef __attribute__((address_space(1))) i4g* gi4p;
typedef __attribute__((address_space(3))) i4g* li4p;

#if defined(__has_builtin)
#if __has_builtin(__builtin_amdgcn_global_load_async_to_lds_b128)
#define HAVE_ASYNC_LDS 1
#endif
#if __has_builtin(__builtin_amdgcn_s_wait_asynccnt)
#define HAVE_WAIT_ASYNC 1
#endif
#endif

__device__ __forceinline__ void wait_async0() {
#if defined(HAVE_WAIT_ASYNC)
  __builtin_amdgcn_s_wait_asynccnt(0);
#elif defined(HAVE_ASYNC_LDS) && defined(__AMDGCN__)
  asm volatile("s_wait_asynccnt 0" ::: "memory");
#endif
}

__device__ __forceinline__ void async_cp16(const _Float16* src, _Float16* dstLds) {
#if defined(HAVE_ASYNC_LDS)
  __builtin_amdgcn_global_load_async_to_lds_b128(
      (gi4p)(uintptr_t)src, (li4p)(uint32_t)(uintptr_t)dstLds, 0, 0);
#else
  *(h8v*)dstLds = *(const h8v*)src;
#endif
}

// Build WMMA A fragment (16x32 f16) for this lane from an LDS row.
// lane<16: elems 0..7 = K kb+0..7,  elems 8..15 = K kb+16..23
// lane>=16: elems 0..7 = K kb+8..15, elems 8..15 = K kb+24..31
__device__ __forceinline__ v16h load_a_frag(const _Float16* sRow, int kb, int sel) {
  h8v lo = *(const h8v*)(sRow + kb + sel);
  h8v hi = *(const h8v*)(sRow + kb + sel + 16);
  v16h a;
#pragma unroll
  for (int i = 0; i < 8; ++i) { a[i] = lo[i]; a[i + 8] = hi[i]; }
  return a;
}

// B fragment: 16 contiguous pre-packed halves per lane.
__device__ __forceinline__ v16h load_b_frag(const _Float16* p) {
  h8v lo = *(const h8v*)(p);
  h8v hi = *(const h8v*)(p + 8);
  v16h b;
#pragma unroll
  for (int i = 0; i < 8; ++i) { b[i] = lo[i]; b[i + 8] = hi[i]; }
  return b;
}

// ---------------------------------------------------------------------------
// Pack W[27][64][64] f32 -> f16 in WMMA B-fragment order:
//   Wp[(((k*4+ntile)*2+kc)*32 + lane)*16 + e]
//   lane<16 : K = kc*32 + e (e<8), kc*32 + 8 + e (e>=8),   N = ntile*16 + lane
//   lane>=16: K = kc*32 + 8 + e (e<8), kc*32 + 16 + e (e>=8), N = ntile*16 + (lane-16)
// Also zero-fills the 64-half "null neighbor" row.
// ---------------------------------------------------------------------------
__global__ void pack_w_kernel(const float* __restrict__ W1,
                              const float* __restrict__ W2,
                              _Float16* __restrict__ W1p,
                              _Float16* __restrict__ W2p,
                              _Float16* __restrict__ zrow) {
  int t = blockIdx.x * blockDim.x + threadIdx.x;
  if (t < CCH) zrow[t] = (_Float16)0.0f;
  if (t >= 2 * K27 * 256) return;
  int which = t / (K27 * 256);
  int u     = t % (K27 * 256);
  const float* W  = which ? W2  : W1;
  _Float16*    Wp = which ? W2p : W1p;

  int lane = u & 31;
  int kc   = (u >> 5) & 1;
  int nt   = (u >> 6) & 3;
  int k    = u >> 8;
  int kbase = kc * 32 + ((lane & 16) ? 8 : 0);
  int ncol  = nt * 16 + (lane & 15);

  _Float16* dst = Wp + (size_t)u * 16;
#pragma unroll
  for (int e = 0; e < 16; ++e) {
    float v = W[((size_t)k * CCH + (kbase + e + (e >= 8 ? 8 : 0))) * CCH + ncol];
    dst[e] = (_Float16)v;
  }
}

// ---------------------------------------------------------------------------
// Stream-convert feats f32 -> f16 (one pass; makes both convs gather 128B rows)
// ---------------------------------------------------------------------------
__global__ __launch_bounds__(256)
void cvt_feats_kernel(const float* __restrict__ feats, _Float16* __restrict__ f16out) {
  size_t i = ((size_t)blockIdx.x * 256 + threadIdx.x) * 8;
  const f4* src = (const f4*)(feats + i);
  f4 a = src[0], b = src[1];
  h8v hv;
  hv[0] = (_Float16)a[0]; hv[1] = (_Float16)a[1];
  hv[2] = (_Float16)a[2]; hv[3] = (_Float16)a[3];
  hv[4] = (_Float16)b[0]; hv[5] = (_Float16)b[1];
  hv[6] = (_Float16)b[2]; hv[7] = (_Float16)b[3];
  *(h8v*)(f16out + i) = hv;
}

// ---------------------------------------------------------------------------
// Fused sparse-conv (27-tap gathered GEMM) + GroupNorm (+ReLU / +residual).
// Block: 32 points, 128 threads = 4 waves; wave w owns output cols [16w,16w+16)
// for BOTH 16-point M-tiles (B fragments reused across M-tiles).
// Gather: f16 rows (srcF = feats16 or h) via async global->LDS, double-buffered.
// phase 1: epilogue GN1+ReLU -> h (f16)
// phase 2: epilogue GN2 + feats residual + ReLU -> out (f32)
// ---------------------------------------------------------------------------
__global__ __launch_bounds__(128)
void spconv_gn_kernel(const _Float16* __restrict__ srcF,   // f16 [N][64] gather source
                      const float* __restrict__ feats,     // f32 residual (phase 2)
                      const int*   __restrict__ nbr,
                      const _Float16* __restrict__ Wp,
                      const float* __restrict__ gamma,
                      const float* __restrict__ beta,
                      _Float16* __restrict__ h,
                      const _Float16* __restrict__ zrow,
                      float* __restrict__ out,
                      int phase) {
  __shared__ _Float16 smemA[2][MTILE * LDA];
  __shared__ float    smemO[MTILE][68];

  const int tid  = threadIdx.x;
  const int lane = tid & 31;
  const int wv   = tid >> 5;            // 0..3: output column tile
  const int pt0  = blockIdx.x * MTILE;

  const int gr = tid >> 3;              // gather row (low 16) 0..15
  const int gc = tid & 7;               // gather chunk of 8 halves

  const int rowA = lane & 15;
  const int selA = (lane & 16) ? 8 : 0;

  v8f acc00 = {}, acc01 = {};           // M-tile 0, kc 0/1
  v8f acc10 = {}, acc11 = {};           // M-tile 1, kc 0/1

  for (int k = 0; k < K27; ++k) {
    const int buf = k & 1;
    _Float16* sbuf = &smemA[buf][0];

    // gather 32 rows x 64 halves: each thread copies 16B for rows gr and gr+16
#pragma unroll
    for (int sub = 0; sub < 2; ++sub) {
      const int row = gr + sub * 16;
      int ni = nbr[(size_t)(pt0 + row) * K27 + k];
      const _Float16* src = (ni >= 0) ? (srcF + (size_t)ni * CCH + gc * 8) : zrow;
      async_cp16(src, sbuf + row * LDA + gc * 8);
    }
    wait_async0();
    __syncthreads();

    const size_t bb = ((((size_t)k * 4 + wv) * 2) * 32 + lane) * 16;
    v16h b0 = load_b_frag(Wp + bb);
    v16h b1 = load_b_frag(Wp + bb + 32 * 16);

    const _Float16* sRow0 = sbuf + rowA * LDA;
    const _Float16* sRow1 = sbuf + (16 + rowA) * LDA;

    v16h a;
    a = load_a_frag(sRow0, 0, selA);
    acc00 = __builtin_amdgcn_wmma_f32_16x16x32_f16(false, a, false, b0, (short)0, acc00, false, false);
    a = load_a_frag(sRow0, 32, selA);
    acc01 = __builtin_amdgcn_wmma_f32_16x16x32_f16(false, a, false, b1, (short)0, acc01, false, false);
    a = load_a_frag(sRow1, 0, selA);
    acc10 = __builtin_amdgcn_wmma_f32_16x16x32_f16(false, a, false, b0, (short)0, acc10, false, false);
    a = load_a_frag(sRow1, 32, selA);
    acc11 = __builtin_amdgcn_wmma_f32_16x16x32_f16(false, a, false, b1, (short)0, acc11, false, false);
  }

  // D layout -> LDS [point][channel]
  const int colD  = wv * 16 + (lane & 15);
  const int mBase = (lane & 16) ? 8 : 0;
#pragma unroll
  for (int i = 0; i < 8; ++i) {
    smemO[mBase + i][colD]      = acc00[i] + acc01[i];
    smemO[16 + mBase + i][colD] = acc10[i] + acc11[i];
  }
  __syncthreads();

  // GroupNorm: 32 points x 8 groups = 256 items on 128 threads (2 iterations)
#pragma unroll
  for (int it = 0; it < 2; ++it) {
    const int idx = tid + it * 128;
    const int r = idx >> 3;
    const int g = idx & 7;
    float x[8], s = 0.f, s2 = 0.f;
#pragma unroll
    for (int j = 0; j < 8; ++j) {
      x[j] = smemO[r][g * 8 + j];
      s  += x[j];
      s2 += x[j] * x[j];
    }
    const float mu  = s * 0.125f;
    const float var = s2 * 0.125f - mu * mu;
    const float rs  = rsqrtf(var + EPSV);
    const int pt = pt0 + r;

    if (phase == 1) {
      h8v hv;
#pragma unroll
      for (int j = 0; j < 8; ++j) {
        int c = g * 8 + j;
        float y = (x[j] - mu) * rs * gamma[c] + beta[c];
        hv[j] = (_Float16)fmaxf(y, 0.0f);
      }
      *(h8v*)(h + (size_t)pt * CCH + g * 8) = hv;
    } else {
      const f4* fr = (const f4*)(feats + (size_t)pt * CCH + g * 8);
      f4 r0 = fr[0], r1 = fr[1];
      f4 o0, o1;
#pragma unroll
      for (int j = 0; j < 4; ++j) {
        int c = g * 8 + j;
        float y = (x[j] - mu) * rs * gamma[c] + beta[c] + r0[j];
        o0[j] = fmaxf(y, 0.0f);
      }
#pragma unroll
      for (int j = 0; j < 4; ++j) {
        int c = g * 8 + 4 + j;
        float y = (x[4 + j] - mu) * rs * gamma[c] + beta[c] + r1[j];
        o1[j] = fmaxf(y, 0.0f);
      }
      f4* po = (f4*)(out + (size_t)pt * CCH + g * 8);
      po[0] = o0;
      po[1] = o1;
    }
  }
}

// ---------------------------------------------------------------------------
extern "C" void kernel_launch(void* const* d_in, const int* in_sizes, int n_in,
                              void* d_out, int out_size, void* d_ws, size_t ws_size,
                              hipStream_t stream) {
  const float* feats = (const float*)d_in[0];
  const float* W1    = (const float*)d_in[1];
  const float* g1    = (const float*)d_in[2];
  const float* b1    = (const float*)d_in[3];
  const float* W2    = (const float*)d_in[4];
  const float* g2    = (const float*)d_in[5];
  const float* b2    = (const float*)d_in[6];
  const int*   nbr   = (const int*)d_in[7];
  float*       out   = (float*)d_out;

  char* ws = (char*)d_ws;
  const size_t WPBYTES = (size_t)K27 * 4 * 2 * 32 * 16 * sizeof(_Float16); // 221184
  const size_t HBYTES  = (size_t)N_PTS * CCH * sizeof(_Float16);           // 51.2 MB
  _Float16* W1p     = (_Float16*)(ws);
  _Float16* W2p     = (_Float16*)(ws + WPBYTES);
  _Float16* zrow    = (_Float16*)(ws + 2 * WPBYTES);
  _Float16* feats16 = (_Float16*)(ws + 2 * WPBYTES + 256);
  _Float16* hbuf    = (_Float16*)(ws + 2 * WPBYTES + 256 + HBYTES);

  (void)in_sizes; (void)n_in; (void)out_size; (void)ws_size;

  // 1) pack weights to WMMA-B layout (+ zero row for missing neighbors)
  const int packThreads = 2 * K27 * 256;                  // 13824
  pack_w_kernel<<<(packThreads + 127) / 128, 128, 0, stream>>>(W1, W2, W1p, W2p, zrow);

  // 2) feats f32 -> f16 (one streaming pass)
  cvt_feats_kernel<<<(N_PTS * CCH) / (256 * 8), 256, 0, stream>>>(feats, feats16);

  // 3) conv1 + GN1 + ReLU -> h (f16)
  spconv_gn_kernel<<<N_PTS / MTILE, 128, 0, stream>>>(feats16, feats, nbr, W1p, g1, b1,
                                                      hbuf, zrow, out, 1);
  // 4) conv2 + GN2 + residual + ReLU -> out (f32)
  spconv_gn_kernel<<<N_PTS / MTILE, 128, 0, stream>>>(hbuf, feats, nbr, W2p, g2, b2,
                                                      hbuf, zrow, out, 2);
}
